// WaveletDiffusion_7894149890684
// MI455X (gfx1250) — matrix operand order repaired and checked by reading, not compiled, not run on hardware
//
#include <hip/hip_runtime.h>

// ---------------------------------------------------------------------------
// Types for CDNA5 WMMA (wave32): A/B = v16h (16 halves = 8 VGPRs), C/D = v8f.
// ---------------------------------------------------------------------------
typedef _Float16 half_t;
typedef _Float16 v16h __attribute__((ext_vector_type(16)));
typedef _Float16 v8h  __attribute__((ext_vector_type(8)));
typedef float    v8f  __attribute__((ext_vector_type(8)));
typedef int      v4i  __attribute__((ext_vector_type(4)));

#if defined(__has_builtin)
#if __has_builtin(__builtin_amdgcn_global_load_async_to_lds_b128)
#define HAVE_ASYNC_LDS 1
#endif
#endif
#ifndef HAVE_ASYNC_LDS
#define HAVE_ASYNC_LDS 0
#endif

// Builtin signature (from clang diagnostic): arg0 = v4i in global AS (printed
// as "v4i __device__ *"), arg1 = LDS pointer, arg2/arg3 = imm offset / cpol.
typedef __attribute__((address_space(1))) v4i* as1_v4ip;
typedef __attribute__((address_space(3))) v4i* as3_v4ip;

#if HAVE_ASYNC_LDS
// 16B global -> LDS copy, no VGPR data movement, tracked by ASYNCcnt.
__device__ __forceinline__ void async_copy16(const half_t* g, half_t* l) {
    __builtin_amdgcn_global_load_async_to_lds_b128(
        (as1_v4ip)(unsigned long long)(const void*)g,
        (as3_v4ip)(unsigned int)(unsigned long long)(void*)l,
        0, 0);
}
#endif

__device__ __forceinline__ void wait_async0() {
#if defined(__has_builtin) && __has_builtin(__builtin_amdgcn_s_wait_asynccnt)
    __builtin_amdgcn_s_wait_asynccnt(0);
#else
    asm volatile("s_wait_asynccnt 0" ::: "memory");
#endif
}

__device__ __forceinline__ v8h zero_v8h() {
    v8h v;
#pragma unroll
    for (int i = 0; i < 8; ++i) v[i] = (half_t)0;
    return v;
}

// ---------------------------------------------------------------------------
// Input conversion: x f32 NCHW (8,8,256,256) -> f16 NHWC [b][y][x][8]
// ---------------------------------------------------------------------------
__global__ void k_convert_in(const float* __restrict__ x, half_t* __restrict__ a, int total) {
    int i = blockIdx.x * 256 + threadIdx.x;
    if (i >= total) return;
    int c = i & 7;
    int p = i >> 3;
    int xx = p & 255;
    int y  = (p >> 8) & 255;
    int b  = p >> 16;
    a[i] = (half_t)x[(((b * 8 + c) * 256) + y) * 256 + xx];
}

// ---------------------------------------------------------------------------
// Weight repack: OIHW f32 -> f16 A-fragment order.
// K-order: k = (ky*3+kx)*CIN + c, chunks of 32, padded with zeros.
// Fragment lane layout (16-bit A 16x32): lane L<16 row=L holds k {0..7,16..23},
// lane L>=16 row=L-16 holds k {8..15,24..31}.
// Packed index: (((kc*(COUT/16) + mtile)*32 + lane)*16 + h)
// ---------------------------------------------------------------------------
__global__ void k_repack(const float* __restrict__ w, half_t* __restrict__ wp,
                         int CIN, int COUT, int NK, int total) {
    int i = blockIdx.x * 256 + threadIdx.x;
    if (i >= total) return;
    int lane = i & 31;
    int t = i >> 5;
    int mtiles = COUT >> 4;
    int mt = t % mtiles;
    int kc = t / mtiles;
    int m = (mt << 4) + (lane & 15);
    int K = 9 * CIN;
#pragma unroll
    for (int h = 0; h < 16; ++h) {
        int koff = ((h < 8) ? h : (h + 8)) + ((lane >> 4) << 3);
        int kg = kc * 32 + koff;
        float v = 0.f;
        if (kg < K) {
            int tap = kg / CIN;
            int c = kg - tap * CIN;
            v = w[(m * CIN + c) * 9 + tap];
        }
        wp[(size_t)i * 16 + h] = (half_t)v;
    }
}

// ---------------------------------------------------------------------------
// Implicit-GEMM 3x3 conv, pad=1, stride STRIDE, NHWC f16 in, WMMA f16->f32.
// Workgroup: 8 waves, tile = (MSUB*16 cout) x (NSUB*64 pixels).
// Each wave: one A-fragment reused across 4 B-fragments (4 v8f accumulators).
// K-loop is software-pipelined: chunk kc+1 is staged into the alternate LDS
// buffer (async-to-LDS when available) while chunk kc runs its 4 WMMAs.
// Epilogue: fused bias (via accumulator init) + group-norm partial stats.
// ---------------------------------------------------------------------------
template <int CIN, int COUT, int STRIDE, bool HAS_GN, bool OUT_F32>
__global__ __launch_bounds__(256) void k_conv_wmma(
    const half_t* __restrict__ act, const half_t* __restrict__ wpack,
    const float* __restrict__ bias,
    half_t* __restrict__ outH, float* __restrict__ outF,
    float* __restrict__ statSum, float* __restrict__ statSq,
    int Hin, int Win, int Hout, int Wout) {
    constexpr int K     = 9 * CIN;
    constexpr int NK    = (K + 31) / 32;
    constexpr int MSUB  = (COUT >= 64) ? 4 : (COUT / 16);
    constexpr int NSUB  = 8 / MSUB;
    constexpr int NTILE = NSUB * 64;
    constexpr int SEGS  = (NTILE * 4) / 256;   // 16B segments per thread per chunk
    constexpr int LBUF  = NTILE * 32;          // halves per LDS buffer

    __shared__ __align__(32) half_t ldsB[2 * LBUF];

    const int tid  = threadIdx.x;
    const int lane = tid & 31;
    const int wave = tid >> 5;
    const int msub = wave / NSUB;
    const int nsub = wave % NSUB;
    const int m0   = (blockIdx.y * MSUB + msub) * 16;
    const int mt   = m0 >> 4;
    const long tileBase = (long)blockIdx.x * NTILE;
    const int npixImg = Hout * Wout;

    // Per-thread im2col pixel precompute (chunk-invariant).
    const int seg = tid & 3;
    int pB[SEGS], pCY[SEGS], pCX[SEGS];
#pragma unroll
    for (int s = 0; s < SEGS; ++s) {
        int pix = s * 64 + (tid >> 2);
        long gp = tileBase + pix;
        int b = (int)(gp / npixImg);
        int r = (int)(gp - (long)b * npixImg);
        int oy = r / Wout;
        int ox = r - oy * Wout;
        pB[s] = b; pCY[s] = oy * STRIDE; pCX[s] = ox * STRIDE;
    }

    // Stage one 32-wide K chunk of the B tile into LDS buffer `buf`.
    auto stage = [&](int kc, int buf) {
        const int k0  = kc * 32 + seg * 8;
        const int tap = k0 / CIN;          // CIN is pow2 -> shifts
        const int c0  = k0 - tap * CIN;
        const int ky  = tap / 3;
        const int kx  = tap - ky * 3;
#pragma unroll
        for (int s = 0; s < SEGS; ++s) {
            half_t* lp = ldsB + buf * LBUF + (s * 64 + (tid >> 2)) * 32 + seg * 8;
            const half_t* gp = nullptr;
            bool inb = false;
            if (k0 < K) {
                int iy = pCY[s] + ky - 1;
                int ix = pCX[s] + kx - 1;
                if (iy >= 0 && iy < Hin && ix >= 0 && ix < Win) {
                    inb = true;
                    gp = act + (((long)(pB[s] * Hin + iy) * Win + ix) * CIN + c0);
                }
            }
            if (inb) {
#if HAVE_ASYNC_LDS
                async_copy16(gp, lp);          // global_load_async_to_lds_b128
#else
                *(v8h*)lp = *(const v8h*)gp;
#endif
            } else {
                *(v8h*)lp = zero_v8h();        // padding / K tail -> zeros
            }
        }
    };

    // Bias-initialized accumulators (GN stats then include bias).
    const int cbase = m0 + ((lane >> 4) << 3);
    v8f acc[4];
#pragma unroll
    for (int r = 0; r < 8; ++r) {
        float bv = bias[cbase + r];
#pragma unroll
        for (int f = 0; f < 4; ++f) acc[f][r] = bv;
    }

    const int mtStride = COUT >> 4;
    stage(0, 0);                               // prologue fill of buffer 0
    for (int kc = 0; kc < NK; ++kc) {
#if HAVE_ASYNC_LDS
        wait_async0();                          // drain own async-to-LDS ops
#endif
        __syncthreads();                        // chunk kc now visible to all

        if (kc + 1 < NK) stage(kc + 1, (kc + 1) & 1);  // overlap with WMMAs

        // A fragment (pre-packed, L2-resident) + next-chunk prefetch
        v16h a = *((const v16h*)wpack + ((long)(kc * mtStride + mt) * 32 + lane));
        if (kc + 1 < NK)
            __builtin_prefetch((const v16h*)wpack +
                               ((long)((kc + 1) * mtStride + mt) * 32 + lane), 0, 3);

        // 4 WMMAs: 16x16x32 f16, f32 accumulate
        const half_t* bbase = ldsB + (kc & 1) * LBUF;
#pragma unroll
        for (int f = 0; f < 4; ++f) {
            const int pixL = nsub * 64 + f * 16 + (lane & 15);
            const half_t* bp = bbase + pixL * 32 + ((lane >> 4) << 3);
            v8h blo = *(const v8h*)bp;
            v8h bhi = *(const v8h*)(bp + 16);
            v16h bfrag = __builtin_shufflevector(blo, bhi, 0, 1, 2, 3, 4, 5, 6, 7, 8, 9, 10, 11, 12, 13, 14, 15);
            acc[f] = __builtin_amdgcn_wmma_f32_16x16x32_f16(
                false, a, false, bfrag, (short)0, acc[f], false, false);
        }
    }

    // ---- fused group-norm partial stats ----
    if constexpr (HAS_GN) {
        float s8[8], q8[8];
#pragma unroll
        for (int r = 0; r < 8; ++r) {
            float s = 0.f, q = 0.f;
#pragma unroll
            for (int f = 0; f < 4; ++f) { float v = acc[f][r]; s += v; q += v * v; }
            s8[r] = s; q8[r] = q;
        }
#pragma unroll
        for (int d = 1; d < 16; d <<= 1) {
#pragma unroll
            for (int r = 0; r < 8; ++r) {
                s8[r] += __shfl_xor(s8[r], d, 32);
                q8[r] += __shfl_xor(q8[r], d, 32);
            }
        }
        if ((lane & 15) == 0) {
            const int b = pB[0];   // pixel tile never crosses a batch boundary
#pragma unroll
            for (int r = 0; r < 8; ++r) {
                atomicAdd(&statSum[b * COUT + cbase + r], s8[r]);
                atomicAdd(&statSq [b * COUT + cbase + r], q8[r]);
            }
        }
    }

    // ---- store NHWC ----
#pragma unroll
    for (int f = 0; f < 4; ++f) {
        const long pixG = tileBase + nsub * 64 + f * 16 + (lane & 15);
        if constexpr (OUT_F32) {
            *(v8f*)(outF + pixG * COUT + cbase) = acc[f];
        } else {
            v8h hv;
#pragma unroll
            for (int r = 0; r < 8; ++r) hv[r] = (half_t)acc[f][r];
            *(v8h*)(outH + pixG * COUT + cbase) = hv;
        }
    }
}

// ---------------------------------------------------------------------------
// GN finalize: per (b,c) scale/shift from per-channel sums (CG channels/group)
// ---------------------------------------------------------------------------
__global__ void k_gn_finalize(const float* __restrict__ sum, const float* __restrict__ sq,
                              const float* __restrict__ gamma, const float* __restrict__ beta,
                              float* __restrict__ sb, int C, int CG, int npix, int BC) {
    int i = blockIdx.x * 256 + threadIdx.x;
    if (i >= BC) return;
    int b = i / C;
    int c = i - b * C;
    int g0 = (c / CG) * CG;
    float S = 0.f, Q = 0.f;
    for (int j = 0; j < CG; ++j) { S += sum[b * C + g0 + j]; Q += sq[b * C + g0 + j]; }
    float cnt  = (float)CG * (float)npix;
    float mean = S / cnt;
    float var  = Q / cnt - mean * mean;
    float inv  = rsqrtf(var + 1e-5f);
    float scale = gamma[c] * inv;
    sb[2 * i]     = scale;
    sb[2 * i + 1] = beta[c] - mean * scale;
}

// ---------------------------------------------------------------------------
// GN apply + ReLU, in place on f16 NHWC, 8 halves per thread
// ---------------------------------------------------------------------------
__global__ void k_gn_apply(half_t* __restrict__ a, const float* __restrict__ sb,
                           int C, int npixImg, long n8) {
    long i = (long)blockIdx.x * 256 + threadIdx.x;
    if (i >= n8) return;
    long off = i * 8;
    long pix = off / C;
    int c0 = (int)(off - pix * C);
    int b  = (int)(pix / npixImg);
    v8h v = *(v8h*)(a + off);
    v8h o;
#pragma unroll
    for (int r = 0; r < 8; ++r) {
        float sc = sb[2 * (b * C + c0 + r)];
        float sh = sb[2 * (b * C + c0 + r) + 1];
        float f  = (float)v[r] * sc + sh;
        o[r] = (half_t)fmaxf(f, 0.f);
    }
    *(v8h*)(a + off) = o;
}

// ---------------------------------------------------------------------------
// Tail: avgpool2(ihaar2(upsample2x(c))) == 0.5 * upsample2x(c[:, ::4])
// c6 is f32 NHWC (8,128,128,16); out is f32 NCHW (8,4,256,256)
// ---------------------------------------------------------------------------
__global__ void k_tail(const float* __restrict__ c6, float* __restrict__ out, int total) {
    int i = blockIdx.x * 256 + threadIdx.x;
    if (i >= total) return;
    int X = i & 255;
    int Y = (i >> 8) & 255;
    int m = (i >> 16) & 3;
    int b = i >> 18;
    int yi = Y >> 1, py = Y & 1;
    int xi = X >> 1, px = X & 1;
    int yo = py ? min(yi + 1, 127) : max(yi - 1, 0);
    int xo = px ? min(xi + 1, 127) : max(xi - 1, 0);
    const int ch = m << 2;
    const float* base = c6 + (long)(b << 7) * 128 * 16 + ch;
    auto V = [&](int y, int x) { return base[(y * 128 + x) * 16]; };
    float v = 0.5625f * V(yi, xi) + 0.1875f * V(yi, xo) +
              0.1875f * V(yo, xi) + 0.0625f * V(yo, xo);
    out[i] = 0.5f * v;
}

// ---------------------------------------------------------------------------
// Host orchestration
// ---------------------------------------------------------------------------
extern "C" void kernel_launch(void* const* d_in, const int* in_sizes, int n_in,
                              void* d_out, int out_size, void* d_ws, size_t ws_size,
                              hipStream_t stream) {
    (void)in_sizes; (void)n_in; (void)out_size; (void)ws_size;
    const float* x   = (const float*)d_in[0];
    const float* w1  = (const float*)d_in[1];  const float* b1  = (const float*)d_in[2];
    const float* g1  = (const float*)d_in[3];  const float* be1 = (const float*)d_in[4];
    const float* w2  = (const float*)d_in[5];  const float* b2  = (const float*)d_in[6];
    const float* g2  = (const float*)d_in[7];  const float* be2 = (const float*)d_in[8];
    const float* w3  = (const float*)d_in[9];  const float* b3  = (const float*)d_in[10];
    const float* g3  = (const float*)d_in[11]; const float* be3 = (const float*)d_in[12];
    const float* w4  = (const float*)d_in[13]; const float* b4  = (const float*)d_in[14];
    const float* g4  = (const float*)d_in[15]; const float* be4 = (const float*)d_in[16];
    const float* w5  = (const float*)d_in[17]; const float* b5  = (const float*)d_in[18];
    const float* g5  = (const float*)d_in[19]; const float* be5 = (const float*)d_in[20];
    const float* w6  = (const float*)d_in[21]; const float* b6  = (const float*)d_in[22];

    char* ws = (char*)d_ws;
    const size_t MB = 1024 * 1024;
    half_t* bufA = (half_t*)(ws);                    // 64 MB ping
    half_t* bufB = (half_t*)(ws + 64 * MB);          // 64 MB pong
    half_t* act0 = (half_t*)(ws + 128 * MB);         // 8 MB f16 input
    float*  c6   = (float*)(ws + 136 * MB);          // 8 MB conv6 f32 out
    half_t* wpB  = (half_t*)(ws + 144 * MB);         // packed weights (~1.5 MB)
    float* statSum = (float*)(ws + 146 * MB);        // 2048 f
    float* statSq  = statSum + 2048;                 // 2048 f
    float* sb      = statSum + 4096;                 // 4096 f (scale,shift)

    half_t* wp1 = wpB;
    half_t* wp2 = wp1 + 6144;       // 64  x 96
    half_t* wp3 = wp2 + 73728;      // 128 x 576
    half_t* wp4 = wp3 + 294912;     // 256 x 1152
    half_t* wp5 = wp4 + 294912;     // 128 x 2304
    half_t* wp6 = wp5 + 73728;      // 64  x 1152   (wp6: 16 x 576)

    // Input -> f16 NHWC
    { int total = 8 * 256 * 256 * 8;
      k_convert_in<<<(total + 255) / 256, 256, 0, stream>>>(x, act0, total); }

    // Repack all weights into WMMA A-fragment order
    auto rp = [&](const float* w, half_t* wp, int CIN, int COUT) {
        int NK = (9 * CIN + 31) / 32;
        int total = NK * (COUT / 16) * 32;
        k_repack<<<(total + 255) / 256, 256, 0, stream>>>(w, wp, CIN, COUT, NK, total);
    };
    rp(w1, wp1, 8, 64);   rp(w2, wp2, 64, 128);  rp(w3, wp3, 128, 256);
    rp(w4, wp4, 256, 128); rp(w5, wp5, 128, 64); rp(w6, wp6, 64, 16);

    // L1: 8 -> 64, 256x256
    (void)hipMemsetAsync(statSum, 0, 4096 * sizeof(float), stream);
    k_conv_wmma<8, 64, 1, true, false><<<dim3(4096, 1), 256, 0, stream>>>(
        act0, wp1, b1, bufA, nullptr, statSum, statSq, 256, 256, 256, 256);
    k_gn_finalize<<<2, 256, 0, stream>>>(statSum, statSq, g1, be1, sb, 64, 8, 65536, 512);
    k_gn_apply<<<16384, 256, 0, stream>>>(bufA, sb, 64, 65536, 4194304L);

    // L2: 64 -> 128, stride 2, 256x256 -> 128x128
    (void)hipMemsetAsync(statSum, 0, 4096 * sizeof(float), stream);
    k_conv_wmma<64, 128, 2, true, false><<<dim3(1024, 2), 256, 0, stream>>>(
        bufA, wp2, b2, bufB, nullptr, statSum, statSq, 256, 256, 128, 128);
    k_gn_finalize<<<4, 256, 0, stream>>>(statSum, statSq, g2, be2, sb, 128, 8, 16384, 1024);
    k_gn_apply<<<8192, 256, 0, stream>>>(bufB, sb, 128, 16384, 2097152L);

    // L3: 128 -> 256, 128x128
    (void)hipMemsetAsync(statSum, 0, 4096 * sizeof(float), stream);
    k_conv_wmma<128, 256, 1, true, false><<<dim3(1024, 4), 256, 0, stream>>>(
        bufB, wp3, b3, bufA, nullptr, statSum, statSq, 128, 128, 128, 128);
    k_gn_finalize<<<8, 256, 0, stream>>>(statSum, statSq, g3, be3, sb, 256, 8, 16384, 2048);
    k_gn_apply<<<16384, 256, 0, stream>>>(bufA, sb, 256, 16384, 4194304L);

    // L4: 256 -> 128, 128x128
    (void)hipMemsetAsync(statSum, 0, 4096 * sizeof(float), stream);
    k_conv_wmma<256, 128, 1, true, false><<<dim3(1024, 2), 256, 0, stream>>>(
        bufA, wp4, b4, bufB, nullptr, statSum, statSq, 128, 128, 128, 128);
    k_gn_finalize<<<4, 256, 0, stream>>>(statSum, statSq, g4, be4, sb, 128, 8, 16384, 1024);
    k_gn_apply<<<8192, 256, 0, stream>>>(bufB, sb, 128, 16384, 2097152L);

    // L5: 128 -> 64, 128x128
    (void)hipMemsetAsync(statSum, 0, 4096 * sizeof(float), stream);
    k_conv_wmma<128, 64, 1, true, false><<<dim3(1024, 1), 256, 0, stream>>>(
        bufB, wp5, b5, bufA, nullptr, statSum, statSq, 128, 128, 128, 128);
    k_gn_finalize<<<2, 256, 0, stream>>>(statSum, statSq, g5, be5, sb, 64, 8, 16384, 512);
    k_gn_apply<<<4096, 256, 0, stream>>>(bufA, sb, 64, 16384, 1048576L);

    // L6: 64 -> 16, 128x128, no GN, f32 out
    k_conv_wmma<64, 16, 1, false, true><<<dim3(256, 1), 256, 0, stream>>>(
        bufA, wp6, b6, nullptr, c6, statSum, statSq, 128, 128, 128, 128);

    // Tail: out = 0.5 * upsample2x(c6[:, ::4])
    k_tail<<<8192, 256, 0, stream>>>(c6, (float*)d_out, 2097152);
}
